// hankel_density_7739531067861
// MI455X (gfx1250) — matrix operand order repaired and checked by reading, not compiled, not run on hardware
//
#include <hip/hip_runtime.h>

// ---------------------------------------------------------------------------
// Problem constants (from reference): B=8192, L=32, D=64, XD=8, R=64, H=256, M=32
// ---------------------------------------------------------------------------
#define LOG2PI 1.8378770664093453f

typedef __attribute__((ext_vector_type(16))) __bf16          v16bf;
typedef __attribute__((ext_vector_type(16))) unsigned short  v16us;
typedef __attribute__((ext_vector_type(8)))  float           v8f;
typedef __attribute__((ext_vector_type(4)))  unsigned int    v4u;
typedef __attribute__((ext_vector_type(8)))  int             v8i;
typedef __attribute__((ext_vector_type(4)))  int             v4i;

#if __has_builtin(__builtin_amdgcn_tensor_load_to_lds) && \
    __has_builtin(__builtin_amdgcn_s_wait_tensorcnt)
#define HAVE_TDM 1
#else
#define HAVE_TDM 0
#endif

// Workspace layout (ushort elements) for bf16-converted weights
#define OFF_CORES 0u        // 32*64*64*64 = 8388608
#define OFF_NW1   8388608u  // 64*256
#define OFF_NW2   8404992u  // 256*64
#define OFF_MUW   8421376u  // 64*256
#define OFF_SIGW  8437760u  // 64*256
#define OFF_AW    8454144u  // 64*32
#define OFF_EW2   8456192u  // 64*64

// LDS carve byte offsets (must match pointer carve in the kernel)
#define LDS_U_BYTE     25600u             // U region
#define LDS_CORE_BYTE  (25600u + 32768u)  // coreL region = 58368

__device__ __forceinline__ unsigned short f2bf(float f) {
  union { float f; unsigned u; } c; c.f = f;
  unsigned r = c.u + 0x7fffu + ((c.u >> 16) & 1u);  // round-to-nearest-even
  return (unsigned short)(r >> 16);
}

// Native transcendentals: v_exp_f32/v_log_f32 are base-2 on AMDGPU.
__device__ __forceinline__ float fexp(float x) {
  return __builtin_amdgcn_exp2f(x * 1.44269504088896f);
}
__device__ __forceinline__ float flog(float x) {
#if __has_builtin(__builtin_amdgcn_logf)
  return __builtin_amdgcn_logf(x) * 0.693147180559945f;
#else
  return __log2f(x) * 0.693147180559945f;
#endif
}
__device__ __forceinline__ float ftanh(float x) {
#if __has_builtin(__builtin_amdgcn_tanhf)
  return __builtin_amdgcn_tanhf(x);
#else
  float e = __builtin_amdgcn_exp2f(x * 2.88539008177793f);  // exp(2x)
  return 1.f - 2.f / (e + 1.f);
#endif
}

__global__ void cvt_bf16_kernel(const float* __restrict__ in,
                                unsigned short* __restrict__ out, int n) {
  int i = blockIdx.x * 256 + threadIdx.x;
  if (i < n) out[i] = f2bf(in[i]);
}

// ---------------------------------------------------------------------------
// WMMA fragment helpers (CDNA5 wave32 layouts, CDNA5 ISA §7.12.2)
// ---------------------------------------------------------------------------
__device__ __forceinline__ v16bf load_a16(const unsigned short* p, int lda,
                                          int rbase, int kbase) {
  int lane = threadIdx.x & 31;
  const unsigned short* rp =
      p + (rbase + (lane & 15)) * lda + kbase + ((lane >> 4) << 3);
  v16us t;
#pragma unroll
  for (int v = 0; v < 8; ++v) {
    unsigned u = *(const unsigned*)(rp + 2 * (v & 3) + ((v >> 2) << 4));
    t[2 * v]     = (unsigned short)(u & 0xffffu);
    t[2 * v + 1] = (unsigned short)(u >> 16);
  }
  return __builtin_bit_cast(v16bf, t);
}

__device__ __forceinline__ v16bf load_b16(const unsigned short* p, int ldb,
                                          int kbase, int cbase) {
  int lane = threadIdx.x & 31;
  const unsigned short* rp =
      p + (kbase + ((lane >> 4) << 3)) * ldb + cbase + (lane & 15);
  v16us t;
#pragma unroll
  for (int j = 0; j < 8; ++j) {
    int kk = 2 * (j & 3) + ((j >> 2) << 4);
    t[2 * j]     = rp[kk * ldb];
    t[2 * j + 1] = rp[(kk + 1) * ldb];
  }
  return __builtin_bit_cast(v16bf, t);
}

__device__ __forceinline__ v8f wmma_bf16(v16bf a, v16bf b, v8f c) {
  return __builtin_amdgcn_wmma_f32_16x16x32_bf16(false, a, false, b, (short)0, c,
                                                 false, false);
}

__device__ __forceinline__ v8f tile_mm(const unsigned short* A, int lda, int rbase,
                                       const unsigned short* B, int ldb, int cbase,
                                       int K) {
  v8f c = {};
  for (int k = 0; k < K; k += 32)
    c = wmma_bf16(load_a16(A, lda, rbase, k), load_b16(B, ldb, k, cbase), c);
  return c;
}

// ---------------------------------------------------------------------------
// Main scan kernel: one block = 32 samples, 128 threads (4 waves), l = 0..31
// Dynamic LDS (125 KB):
//   h[32][64] f32 | hn[32][64] f32 | e[32][64] f32 | x[32][8] f32 | U (100 KB)
// U (phase-union):
//   MDN   : shbf | z1bf | z2bf | e1bf (bf16) ; muL | sigL | alL (f32)
//   einsum: G[32][512] bf16 | coreL[512][64] bf16 (TDM-staged)
// ---------------------------------------------------------------------------
__global__ __launch_bounds__(128, 1)
void hankel_main_kernel(const float* __restrict__ X,
                        const float* __restrict__ init_w,
                        const float* __restrict__ ew1,
                        const float* __restrict__ eb1,
                        const float* __restrict__ eb2,
                        const float* __restrict__ nb1,
                        const float* __restrict__ nb2,
                        const float* __restrict__ mu_b,
                        const float* __restrict__ sig_b,
                        const float* __restrict__ a_b,
                        const unsigned short* __restrict__ ws,
                        float* __restrict__ out) {
  extern __shared__ __align__(16) char smem[];
  float* h  = (float*)smem;
  float* hn = h  + 2048;
  float* eF = hn + 2048;
  float* xb = eF + 2048;
  unsigned short* U = (unsigned short*)(xb + 256);  // byte offset 25600

  unsigned short* shbf = U;
  unsigned short* z1bf = U + 2048;
  unsigned short* z2bf = U + 10240;
  unsigned short* e1bf = U + 12288;
  float* muL  = (float*)(U + 16384);
  float* sigL = muL + 8192;
  float* alL  = sigL + 8192;
  unsigned short* G     = U;
  unsigned short* coreL = U + 16384;  // byte offset 58368 == LDS_CORE_BYTE

  const unsigned short* coresB = ws + OFF_CORES;
  const unsigned short* nw1B   = ws + OFF_NW1;
  const unsigned short* nw2B   = ws + OFF_NW2;
  const unsigned short* muwB   = ws + OFF_MUW;
  const unsigned short* sigwB  = ws + OFF_SIGW;
  const unsigned short* awB    = ws + OFF_AW;
  const unsigned short* ew2B   = ws + OFF_EW2;

  const int tid  = threadIdx.x;
  const int wv   = tid >> 5;
  const int lane = tid & 31;
  const int colL = lane & 15;
  const int grp  = lane >> 4;
  const int n0   = blockIdx.x * 32;

  for (int i = tid; i < 2048; i += 128) h[i] = init_w[i & 63];
  float ll_acc = 0.f, nrm_acc = 0.f;
  __syncthreads();

  for (int l = 0; l < 32; ++l) {
    // ---- P0: load x tile; softmax(h) + ||h||^2 (one sample per lane-thread)
    for (int i = tid; i < 256; i += 128) {
      int s = i >> 3, q = i & 7;
      xb[i] = X[(n0 + s) * 256 + q * 32 + l];  // X[n, xd, l]
    }
    if (tid < 32) {
      const float* hr = h + tid * 64;
      float mx = -1e30f, nv = 0.f;
      for (int j = 0; j < 64; ++j) { float v = hr[j]; nv += v * v; mx = fmaxf(mx, v); }
      float sum = 0.f;
      for (int j = 0; j < 64; ++j) sum += fexp(hr[j] - mx);
      float inv = 1.f / sum;
      for (int j = 0; j < 64; ++j)
        shbf[tid * 64 + j] = f2bf(fexp(hr[j] - mx) * inv);
      nrm_acc += nv;
    }
    __syncthreads();

    // ---- P1: z1 = tanh(sh @ nw1 + nb1)   [32x64]@[64x256]
    for (int idx = wv; idx < 32; idx += 4) {
      int rt = idx >> 4, ct = idx & 15;
      v8f c = tile_mm(shbf, 64, rt * 16, nw1B, 256, ct * 16, 64);
      int col = ct * 16 + colL;
      float bb = nb1[col];
#pragma unroll
      for (int r = 0; r < 8; ++r)
        z1bf[(rt * 16 + r + 8 * grp) * 256 + col] = f2bf(ftanh(c[r] + bb));
    }
    __syncthreads();

    // ---- P2: z2 = tanh(z1 @ nw2 + nb2)   [32x256]@[256x64]
    for (int idx = wv; idx < 8; idx += 4) {
      int rt = idx >> 2, ct = idx & 3;
      v8f c = tile_mm(z1bf, 256, rt * 16, nw2B, 64, ct * 16, 256);
      int col = ct * 16 + colL;
      float bb = nb2[col];
#pragma unroll
      for (int r = 0; r < 8; ++r)
        z2bf[(rt * 16 + r + 8 * grp) * 64 + col] = f2bf(ftanh(c[r] + bb));
    }
    __syncthreads();

    // ---- P3: mu/sig/alpha linears; encoder layer 1 on lane-threads
    for (int idx = wv; idx < 32; idx += 4) {
      int rt = idx >> 4, ct = idx & 15;
      v8f c = tile_mm(z2bf, 64, rt * 16, muwB, 256, ct * 16, 64);
      int col = ct * 16 + colL; float bb = mu_b[col];
#pragma unroll
      for (int r = 0; r < 8; ++r)
        muL[(rt * 16 + r + 8 * grp) * 256 + col] = c[r] + bb;
    }
    for (int idx = wv; idx < 32; idx += 4) {
      int rt = idx >> 4, ct = idx & 15;
      v8f c = tile_mm(z2bf, 64, rt * 16, sigwB, 256, ct * 16, 64);
      int col = ct * 16 + colL; float bb = sig_b[col];
#pragma unroll
      for (int r = 0; r < 8; ++r)
        sigL[(rt * 16 + r + 8 * grp) * 256 + col] = c[r] + bb;
    }
    for (int idx = wv; idx < 4; idx += 4) {
      int rt = idx >> 1, ct = idx & 1;
      v8f c = tile_mm(z2bf, 64, rt * 16, awB, 32, ct * 16, 64);
      int col = ct * 16 + colL; float bb = a_b[col];
#pragma unroll
      for (int r = 0; r < 8; ++r)
        alL[(rt * 16 + r + 8 * grp) * 32 + col] = c[r] + bb;
    }
    if (tid < 32) {  // e1 = relu(x @ ew1 + eb1): tiny (K=8), fp32 VALU
      const float* xr = xb + tid * 8;
      for (int d = 0; d < 64; ++d) {
        float acc = eb1[d];
#pragma unroll
        for (int q = 0; q < 8; ++q) acc += xr[q] * ew1[q * 64 + d];
        e1bf[tid * 64 + d] = f2bf(fmaxf(acc, 0.f));
      }
    }
    __syncthreads();

    // ---- P4: e = e1 @ ew2 + eb2 ; mixture log-density (phi) on lane-threads
    for (int idx = wv; idx < 8; idx += 4) {
      int rt = idx >> 2, ct = idx & 3;
      v8f c = tile_mm(e1bf, 64, rt * 16, ew2B, 64, ct * 16, 64);
      int col = ct * 16 + colL; float bb = eb2[col];
#pragma unroll
      for (int r = 0; r < 8; ++r)
        eF[(rt * 16 + r + 8 * grp) * 64 + col] = c[r] + bb;
    }
    if (tid < 32) {
      const float* mr = muL + tid * 256;
      const float* sr = sigL + tid * 256;
      const float* ar = alL + tid * 32;
      const float* xr = xb + tid * 8;
      float amax = -1e30f;
#pragma unroll
      for (int m = 0; m < 32; ++m) amax = fmaxf(amax, ar[m]);
      float asum = 0.f;
#pragma unroll
      for (int m = 0; m < 32; ++m) asum += fexp(ar[m] - amax);
      float logZ = amax + flog(asum);
      float tvals[32];
      float tmax = -1e30f;
#pragma unroll
      for (int m = 0; m < 32; ++m) {
        float acc = -0.5f * 8.0f * LOG2PI;
#pragma unroll
        for (int q = 0; q < 8; ++q) {
          float sv = sr[m * 8 + q];
          float d  = (xr[q] - mr[m * 8 + q]) * fexp(-sv);
          acc += -0.5f * d * d - sv;
        }
        float tv = ar[m] - logZ + acc;
        tvals[m] = tv;
        tmax = fmaxf(tmax, tv);
      }
      float tsum = 0.f;
#pragma unroll
      for (int m = 0; m < 32; ++m) tsum += fexp(tvals[m] - tmax);
      ll_acc += tmax + flog(tsum);
    }
    __syncthreads();

    // ---- P5: h_next = (h (x) e) @ core_l   [32x4096]@[4096x64], K-chunks of 512
    v8f acc0 = {}, acc1 = {};
    const int rtE = wv >> 1;
    const int cbE = (wv & 1) * 32;
    for (int ch = 0; ch < 8; ++ch) {
      const int kc = ch * 512;
      const unsigned short* src = coresB + (size_t)l * 262144 + (size_t)kc * 64;

#if HAVE_TDM
      // Tensor Data Mover: DMA the 512x64 bf16 core chunk into LDS (wave 0
      // issues; TDM ignores EXEC and signals completion once). The G-build
      // below overlaps with the DMA; s_wait_tensorcnt + barrier close it.
      if (wv == 0) {
        unsigned long long ga = (unsigned long long)(const void*)src;
        v4u g0;
        g0[0] = 1u;                                    // count=1 valid descriptor
        g0[1] = LDS_CORE_BYTE;                         // lds_addr
        g0[2] = (unsigned)ga;                          // global_addr[31:0]
        g0[3] = (unsigned)((ga >> 32) & 0x01FFFFFFu)   // global_addr[56:32]
                | 0x80000000u;                         // type=2 ("image")
        v8i g1;
        g1[0] = (int)(1u << 16);     // workgroup_mask=0, data_size=1 (2 bytes)
        g1[1] = (int)(64u << 16);    // tensor_dim0 = 64
        g1[2] = (int)(512u << 16);   // tensor_dim1 = 512
        g1[3] = (int)(64u << 16);    // tile_dim0 = 64
        g1[4] = (int)512;            // tile_dim1 = 512
        g1[5] = (int)64;             // tensor_dim0_stride = 64
        g1[6] = 0;
        g1[7] = 0;
        v4i gz4 = {0, 0, 0, 0};
        v8i gz8 = {0, 0, 0, 0, 0, 0, 0, 0};
        // clang-23 (amdgpu-toolchain) 6-arg form:
        // (uint32x4 g0, int32x8 g1, int32x4, int32x4, int32x8, i32 cpol)
        __builtin_amdgcn_tensor_load_to_lds(g0, g1, gz4, gz4, gz8, 0);
      }
#else
      {  // fallback: staged copy through VGPRs
        const uint4* s4 = (const uint4*)src;
        uint4* d4 = (uint4*)coreL;
        for (int i = tid; i < 4096; i += 128) d4[i] = s4[i];
      }
#endif
      if (ch + 1 < 8)  // global_prefetch of next chunk (L2-resident set)
        __builtin_prefetch(src + 32768 + tid * 256, 0, 0);

      // build G chunk: G[s][k] = h[s][ (kc+k)>>6 ] * e[s][ (kc+k)&63 ]
      for (int p = tid; p < 8192; p += 128) {
        int s = p >> 8, pr = p & 255;
        int gk = kc + pr * 2;
        float f0 = h[s * 64 + (gk >> 6)] * eF[s * 64 + (gk & 63)];
        float f1 = h[s * 64 + ((gk + 1) >> 6)] * eF[s * 64 + ((gk + 1) & 63)];
        ((unsigned*)G)[s * 256 + pr] =
            (unsigned)f2bf(f0) | ((unsigned)f2bf(f1) << 16);
      }
#if HAVE_TDM
      __builtin_amdgcn_s_wait_tensorcnt(0);  // no-op for waves 1..3
#endif
      __syncthreads();

#pragma unroll 4
      for (int kt = 0; kt < 16; ++kt) {
        v16bf a  = load_a16(G, 512, rtE * 16, kt * 32);
        v16bf b0 = load_b16(coreL, 64, kt * 32, cbE);
        v16bf b1 = load_b16(coreL, 64, kt * 32, cbE + 16);
        acc0 = wmma_bf16(a, b0, acc0);
        acc1 = wmma_bf16(a, b1, acc1);
      }
      __syncthreads();
    }
    // ---- P6: commit h_next, rotate state
#pragma unroll
    for (int r = 0; r < 8; ++r) {
      int row = rtE * 16 + r + 8 * grp;
      hn[row * 64 + cbE + colL]      = acc0[r];
      hn[row * 64 + cbE + 16 + colL] = acc1[r];
    }
    __syncthreads();
    for (int i = tid; i < 2048; i += 128) h[i] = hn[i];
    __syncthreads();
  }

  if (tid < 32) {
    out[n0 + tid]        = ll_acc;   // sum_l log-likelihood
    out[8192 + n0 + tid] = nrm_acc;  // sum_l ||h||^2
  }
}

// ---------------------------------------------------------------------------
extern "C" void kernel_launch(void* const* d_in, const int* in_sizes, int n_in,
                              void* d_out, int out_size, void* d_ws, size_t ws_size,
                              hipStream_t stream) {
  const float* X      = (const float*)d_in[0];
  const float* init_w = (const float*)d_in[1];
  const float* cores  = (const float*)d_in[2];
  const float* ew1    = (const float*)d_in[3];
  const float* eb1    = (const float*)d_in[4];
  const float* ew2    = (const float*)d_in[5];
  const float* eb2    = (const float*)d_in[6];
  const float* nw1    = (const float*)d_in[7];
  const float* nb1    = (const float*)d_in[8];
  const float* nw2    = (const float*)d_in[9];
  const float* nb2    = (const float*)d_in[10];
  const float* mu_w   = (const float*)d_in[11];
  const float* mu_b   = (const float*)d_in[12];
  const float* sig_w  = (const float*)d_in[13];
  const float* sig_b  = (const float*)d_in[14];
  const float* a_w    = (const float*)d_in[15];
  const float* a_b    = (const float*)d_in[16];
  unsigned short* ws  = (unsigned short*)d_ws;
  float* out = (float*)d_out;

  auto cvt = [&](const float* p, unsigned off, int n) {
    cvt_bf16_kernel<<<(n + 255) / 256, 256, 0, stream>>>(p, ws + off, n);
  };
  cvt(cores, OFF_CORES, 8388608);
  cvt(nw1,   OFF_NW1,   16384);
  cvt(nw2,   OFF_NW2,   16384);
  cvt(mu_w,  OFF_MUW,   16384);
  cvt(sig_w, OFF_SIGW,  16384);
  cvt(a_w,   OFF_AW,    2048);
  cvt(ew2,   OFF_EW2,   4096);

  const size_t smem_bytes = (3 * 2048 + 256) * 4 + 51200 * 2;  // 128000 B
  hankel_main_kernel<<<256, 128, smem_bytes, stream>>>(
      X, init_w, ew1, eb1, eb2, nb1, nb2, mu_b, sig_b, a_b, ws, out);
}
